// GNN_Block_30777735643935
// MI455X (gfx1250) — matrix operand order repaired and checked by reading, not compiled, not run on hardware
//
#include <hip/hip_runtime.h>
#include <hip/hip_bf16.h>
#include <stdint.h>
#include <stddef.h>

#define HID 128
#define MT 4   // M-subtiles (16 rows each) per wave, sharing B fragments

typedef __attribute__((ext_vector_type(16))) _Float16 v16h;
typedef __attribute__((ext_vector_type(8)))  float    v8f;

union HalfPack { v16h h; uint32_t u[8]; };

// ---------------------------------------------------------------------------
// WMMA fragment loaders (wave32, 16x16x32 f16 -> f32)
//
// A (16x32, f16): lane L<16 holds row M=L; VGPR v holds K pair
//   kb(v) = 2v + 16*(L>=16)    (contiguous, matches B)
// B (32x16, f16) from pre-transposed Wt[N][K]: lane L<16 holds col N=L with
//   K=0..15 packed 2-per-VGPR; lanes 16..31 hold K=16..31.
// ---------------------------------------------------------------------------
__device__ __forceinline__ v16h load_a_frag(const _Float16* A, int row0, int k0,
                                            int lane, int M) {
  int row = row0 + (lane & 15);
  if (row >= M) row = M - 1;  // clamp only matters in the last partial tile
  const int hi = (lane >> 4) & 1;
  const uint32_t* p = (const uint32_t*)(A + (size_t)row * HID + k0);
  HalfPack f;
#pragma unroll
  for (int v = 0; v < 8; ++v) {
    const int kb = 2 * v + 16 * hi;
    f.u[v] = p[kb >> 1];
  }
  return f.h;
}

__device__ __forceinline__ v16h load_b_frag(const _Float16* Wt, int col0, int k0,
                                            int lane) {
  const int col = col0 + (lane & 15);
  const int hi = (lane >> 4) & 1;
  const uint32_t* p = (const uint32_t*)(Wt + (size_t)col * HID + k0 + 16 * hi);
  HalfPack f;
#pragma unroll
  for (int v = 0; v < 8; ++v) f.u[v] = p[v];
  return f.h;
}

// ---------------------------------------------------------------------------
// Prep kernels
// ---------------------------------------------------------------------------
__global__ void zero_f32(float* p, int n) {
  const int i = blockIdx.x * blockDim.x + threadIdx.x;
  if (i < n) p[i] = 0.0f;
}

__global__ void cvt_f32_to_f16(const float* in, _Float16* out, int n) {
  const int i = blockIdx.x * blockDim.x + threadIdx.x;
  if (i < n) out[i] = (_Float16)in[i];
}

// W[k][n] f32 (128x128) -> Wt[n][k] f16
__global__ void transpose_cvt_128(const float* W, _Float16* Wt) {
  const int i = blockIdx.x * blockDim.x + threadIdx.x;  // 16384 threads
  const int k = i >> 7, n = i & 127;
  Wt[(size_t)n * HID + k] = (_Float16)W[(size_t)k * HID + n];
}

__global__ void degree_count(const int* dst, float* deg, int E) {
  const int e = blockIdx.x * blockDim.x + threadIdx.x;
  if (e < E) atomicAdd(&deg[dst[e]], 1.0f);
}

// in place: deg -> rsqrt(deg + 1 self-loop); deg+1 >= 1 so always finite
__global__ void make_dinv(float* d, int n) {
  const int i = blockIdx.x * blockDim.x + threadIdx.x;
  if (i < n) d[i] = rsqrtf(d[i] + 1.0f);
}

// ---------------------------------------------------------------------------
// GEMM: Out_f16[M,128] = A_f16[M,128] @ W  (W given transposed f16 [128][128])
// One wave = 64x16 C (4 subtiles of 16x16 sharing each B fragment);
// 8 waves/block cover N=128; grid.x covers ceil(M/64).
// ---------------------------------------------------------------------------
__global__ __launch_bounds__(256) void gemm_h16(const _Float16* __restrict__ A,
                                                const _Float16* __restrict__ Wt,
                                                _Float16* __restrict__ Out, int M) {
  const int block_m = blockIdx.x * (16 * MT);
  const int wave = threadIdx.x >> 5;
  const int lane = threadIdx.x & 31;
  const int ncol = wave * 16;
  v8f c[MT];
#pragma unroll
  for (int t = 0; t < MT; ++t) c[t] = (v8f){};

#pragma unroll
  for (int k0 = 0; k0 < HID; k0 += 32) {
    const v16h b = load_b_frag(Wt, ncol, k0, lane);
#pragma unroll
    for (int t = 0; t < MT; ++t) {
      const v16h a = load_a_frag(A, block_m + 16 * t, k0, lane, M);
      c[t] = __builtin_amdgcn_wmma_f32_16x16x32_f16(false, a, false, b,
                                                    (short)0, c[t], false, false);
    }
  }

  const int col = ncol + (lane & 15);
  const int hi = (lane >> 4) & 1;
#pragma unroll
  for (int t = 0; t < MT; ++t) {
    const int tile_m = block_m + 16 * t;
    if (tile_m + 16 <= M) {           // uniform branch: straight-line stores
#pragma unroll
      for (int r = 0; r < 8; ++r)
        Out[(size_t)(tile_m + r + 8 * hi) * HID + col] = (_Float16)c[t][r];
    } else if (tile_m < M) {          // rare partial tile
#pragma unroll
      for (int r = 0; r < 8; ++r) {
        const int row = tile_m + r + 8 * hi;
        if (row < M) Out[(size_t)row * HID + col] = (_Float16)c[t][r];
      }
    }
  }
}

// ---------------------------------------------------------------------------
// Edge scatter: one wave per edge; agg[dst] += norm * h[src]   (f32 atomics,
// hot set is L2-resident: h=12.8MB f16, agg=25.6MB f32 << 192MB L2)
// ---------------------------------------------------------------------------
__global__ __launch_bounds__(256) void edge_scatter(const _Float16* __restrict__ H,
                                                    const int* __restrict__ src,
                                                    const int* __restrict__ dst,
                                                    const float* __restrict__ dinv,
                                                    float* __restrict__ Agg, int E) {
  const int e = blockIdx.x * 8 + (threadIdx.x >> 5);
  if (e >= E) return;
  const int lane = threadIdx.x & 31;
  const int u = src[e];
  const int v = dst[e];
  const float norm = dinv[u] * dinv[v];
  const uint32_t* hp = (const uint32_t*)(H + (size_t)u * HID) + lane * 2;
  float* ap = Agg + (size_t)v * HID + lane * 4;
  union { uint32_t u32; _Float16 h[2]; } c0, c1;
  c0.u32 = hp[0];
  c1.u32 = hp[1];
  atomicAdd(ap + 0, norm * (float)c0.h[0]);
  atomicAdd(ap + 1, norm * (float)c0.h[1]);
  atomicAdd(ap + 2, norm * (float)c1.h[0]);
  atomicAdd(ap + 3, norm * (float)c1.h[1]);
}

// Self-loop term + bias + ReLU, emit f16 for next GEMM stage
__global__ void finish_relu(const float* __restrict__ Agg,
                            const _Float16* __restrict__ H,
                            const float* __restrict__ dinv,
                            const float* __restrict__ bias,
                            _Float16* __restrict__ Xout, int total) {
  const int i = blockIdx.x * blockDim.x + threadIdx.x;
  if (i >= total) return;
  const int node = i >> 7;
  const int f = i & 127;
  const float dv = dinv[node];
  const float val = Agg[i] + dv * dv * (float)H[i] + bias[f];
  Xout[i] = (_Float16)fmaxf(val, 0.0f);
}

// ---------------------------------------------------------------------------
// Head: Out_f32 = X1 @ WlinTop + X2 @ WlinBot + blin (concat GEMM fused).
// Same 4-subtile structure; B1/B2 fragments shared across subtiles.
// ---------------------------------------------------------------------------
__global__ __launch_bounds__(256) void gemm_concat_final(
    const _Float16* __restrict__ X1, const _Float16* __restrict__ X2,
    const _Float16* __restrict__ WtA, const _Float16* __restrict__ WtB,
    const float* __restrict__ bias, float* __restrict__ Out, int M) {
  const int block_m = blockIdx.x * (16 * MT);
  const int wave = threadIdx.x >> 5;
  const int lane = threadIdx.x & 31;
  const int ncol = wave * 16;
  v8f c[MT];
#pragma unroll
  for (int t = 0; t < MT; ++t) c[t] = (v8f){};

#pragma unroll
  for (int k0 = 0; k0 < HID; k0 += 32) {
    const v16h b1 = load_b_frag(WtA, ncol, k0, lane);
#pragma unroll
    for (int t = 0; t < MT; ++t) {
      const v16h a1 = load_a_frag(X1, block_m + 16 * t, k0, lane, M);
      c[t] = __builtin_amdgcn_wmma_f32_16x16x32_f16(false, a1, false, b1,
                                                    (short)0, c[t], false, false);
    }
    const v16h b2 = load_b_frag(WtB, ncol, k0, lane);
#pragma unroll
    for (int t = 0; t < MT; ++t) {
      const v16h a2 = load_a_frag(X2, block_m + 16 * t, k0, lane, M);
      c[t] = __builtin_amdgcn_wmma_f32_16x16x32_f16(false, a2, false, b2,
                                                    (short)0, c[t], false, false);
    }
  }

  const int col = ncol + (lane & 15);
  const int hi = (lane >> 4) & 1;
  const float bv = bias[col];
#pragma unroll
  for (int t = 0; t < MT; ++t) {
    const int tile_m = block_m + 16 * t;
    if (tile_m + 16 <= M) {
#pragma unroll
      for (int r = 0; r < 8; ++r)
        Out[(size_t)(tile_m + r + 8 * hi) * HID + col] = c[t][r] + bv;
    } else if (tile_m < M) {
#pragma unroll
      for (int r = 0; r < 8; ++r) {
        const int row = tile_m + r + 8 * hi;
        if (row < M) Out[(size_t)row * HID + col] = c[t][r] + bv;
      }
    }
  }
}

// ---------------------------------------------------------------------------
extern "C" void kernel_launch(void* const* d_in, const int* in_sizes, int n_in,
                              void* d_out, int out_size, void* d_ws, size_t ws_size,
                              hipStream_t stream) {
  const float* x     = (const float*)d_in[0];
  const int*   ei    = (const int*)d_in[1];      // [2, E] flat
  const float* W1    = (const float*)d_in[2];
  const float* b1    = (const float*)d_in[3];
  const float* W2    = (const float*)d_in[4];
  const float* b2    = (const float*)d_in[5];
  const float* Wlin  = (const float*)d_in[6];    // [256, 128]
  const float* blin  = (const float*)d_in[7];
  float*       out   = (float*)d_out;

  const int N = in_sizes[0] / HID;   // 50000
  const int E = in_sizes[1] / 2;     // 640000
  const int* src = ei;
  const int* dst = ei + E;

  // Carve scratch (256B aligned)
  char* ws = (char*)d_ws;
  size_t off = 0;
  auto carve = [&](size_t bytes) -> char* {
    char* p = ws + off;
    off = (off + bytes + 255) & ~(size_t)255;
    return p;
  };
  float*    dinv = (float*)carve((size_t)N * 4);
  _Float16* xh   = (_Float16*)carve((size_t)N * HID * 2);
  _Float16* hh   = (_Float16*)carve((size_t)N * HID * 2);
  _Float16* x1h  = (_Float16*)carve((size_t)N * HID * 2);
  _Float16* x2h  = (_Float16*)carve((size_t)N * HID * 2);
  float*    agg  = (float*)carve((size_t)N * HID * 4);
  _Float16* w1t  = (_Float16*)carve((size_t)HID * HID * 2);
  _Float16* w2t  = (_Float16*)carve((size_t)HID * HID * 2);
  _Float16* wlt  = (_Float16*)carve((size_t)HID * HID * 2);
  _Float16* wlb  = (_Float16*)carve((size_t)HID * HID * 2);
  (void)ws_size; (void)n_in; (void)out_size;

  const int feat_total = N * HID;
  const int gB = 256;
  const int gFeat  = (feat_total + gB - 1) / gB;
  const int gNode  = (N + gB - 1) / gB;
  const int gEdge  = (E + gB - 1) / gB;
  const int gScat  = (E + 7) / 8;
  const int gTiles = (N + 16 * MT - 1) / (16 * MT);

  // 1) degrees -> dinv (in place)
  zero_f32<<<gNode, gB, 0, stream>>>(dinv, N);
  degree_count<<<gEdge, gB, 0, stream>>>(dst, dinv, E);
  make_dinv<<<gNode, gB, 0, stream>>>(dinv, N);

  // 2) precision/layout prep
  cvt_f32_to_f16<<<gFeat, gB, 0, stream>>>(x, xh, feat_total);
  transpose_cvt_128<<<64, gB, 0, stream>>>(W1, w1t);
  transpose_cvt_128<<<64, gB, 0, stream>>>(W2, w2t);
  transpose_cvt_128<<<64, gB, 0, stream>>>(Wlin, wlt);               // rows 0..127
  transpose_cvt_128<<<64, gB, 0, stream>>>(Wlin + HID * HID, wlb);   // rows 128..255

  // 3) layer 1: h = x@W1 ; aggregate ; relu
  gemm_h16<<<gTiles, gB, 0, stream>>>(xh, w1t, hh, N);
  zero_f32<<<gFeat, gB, 0, stream>>>(agg, feat_total);
  edge_scatter<<<gScat, gB, 0, stream>>>(hh, src, dst, dinv, agg, E);
  finish_relu<<<gFeat, gB, 0, stream>>>(agg, hh, dinv, b1, x1h, feat_total);

  // 4) layer 2: h = x1@W2 ; aggregate ; relu
  gemm_h16<<<gTiles, gB, 0, stream>>>(x1h, w2t, hh, N);
  zero_f32<<<gFeat, gB, 0, stream>>>(agg, feat_total);
  edge_scatter<<<gScat, gB, 0, stream>>>(hh, src, dst, dinv, agg, E);
  finish_relu<<<gFeat, gB, 0, stream>>>(agg, hh, dinv, b2, x2h, feat_total);

  // 5) head: out = concat(x1,x2) @ Wlin + blin  (fused two-operand GEMM)
  gemm_concat_final<<<gTiles, gB, 0, stream>>>(x1h, x2h, wlt, wlb, blin, out, N);
}
